// TensorProductConvLayer_25589415149822
// MI455X (gfx1250) — matrix operand order repaired and checked by reading, not compiled, not run on hardware
//
#include <hip/hip_runtime.h>

typedef __attribute__((ext_vector_type(16))) _Float16 v16h;
typedef __attribute__((ext_vector_type(8)))  float    v8f;

#define DIN 64
#define NEF 32

__device__ __forceinline__ v16h lds_load16h(const _Float16* p) {
    v16h r;
    __builtin_memcpy(&r, p, 32);   // 32B aligned -> 2x ds_load_b128
    return r;
}

__global__ __launch_bounds__(128)
void tpconv_fused_kernel(const float* __restrict__ node_attr,
                         const float* __restrict__ edge_attr,
                         const float* __restrict__ edge_sh,
                         const float* __restrict__ mask,
                         const float* __restrict__ w_in_s,
                         const float* __restrict__ w_in_v,
                         const float* __restrict__ fc_w1,
                         const float* __restrict__ fc_b1,
                         const float* __restrict__ fc_w2,
                         const float* __restrict__ fc_b2,
                         const float* __restrict__ w_out_s,
                         const float* __restrict__ w_out_v,
                         float* __restrict__ out,
                         int N)
{
    __shared__ _Float16 Mt[128 * 32];      // fused per-node matrix, col-major (col*32+16*(k/8%2)+8*(k/16)+k%8)
    __shared__ float Cvec[128];            // fc_b2-derived per-node bias on y
    __shared__ float na_sh[64];
    __shared__ float sin_sh[16];
    __shared__ float vin_sh[48];
    __shared__ float fb1_sh[32];
    __shared__ float redS[128];
    __shared__ float redV[384];
    __shared__ float Ssum[16];
    __shared__ float Vsum[48];
    __shared__ float dsum_sh;

    const int tid  = threadIdx.x;
    const int node = blockIdx.x;        // node = b*N + n
    const int b    = node / N;
    const int n    = node % N;

    // ---------------- Phase A: per-node setup ----------------
    if (tid < 64) na_sh[tid] = node_attr[(size_t)node * DIN + tid];
    {
        float p = 0.f;
        for (int i = tid; i < N; i += 128) p += mask[(size_t)b * N + i];
        redS[tid] = p;
    }
    __syncthreads();

    if (tid < 16) {
        float a = 0.f;
        #pragma unroll
        for (int j = 0; j < 16; ++j) a += na_sh[j] * w_in_s[j * 16 + tid];
        sin_sh[tid] = a * 0.25f;                       // /sqrt(16)
    } else if (tid < 48) {
        fb1_sh[tid - 16] = fc_b1[tid - 16];
    } else if (tid >= 64 && tid < 112) {
        int idx = tid - 64; int o = idx / 3; int x = idx % 3;
        float a = 0.f;
        #pragma unroll
        for (int j = 0; j < 16; ++j) a += na_sh[16 + j * 3 + x] * w_in_v[j * 16 + o];
        vin_sh[o * 3 + x] = a * 0.25f;
    }
    if (tid == 127) {
        float s = 0.f;
        for (int i = 0; i < 128; ++i) s += redS[i];
        dsum_sh = s - 1.0f;
    }
    __syncthreads();

    // Build fused 32x128 matrix Mt and bias Cvec. Column blocks:
    //  [0,16): A_ss (w off 0)   [16,32): A_sv (off 512)
    //  [32,80): A_vs x-major (off 768)   [80,128): B_vv x-major (off 256)
    {
        const int col = tid;
        int off, o, x = 0;
        bool use_s;
        if (col < 16)      { o = col;      off = 0;   use_s = true;  }
        else if (col < 32) { o = col - 16; off = 512; use_s = true;  }
        else if (col < 80) { int i2 = col - 32; x = i2 >> 4; o = i2 & 15; off = 768; use_s = false; }
        else               { int i2 = col - 80; x = i2 >> 4; o = i2 & 15; off = 256; use_s = false; }
        float src[16];
        #pragma unroll
        for (int i = 0; i < 16; ++i) src[i] = use_s ? sin_sh[i] : vin_sh[i * 3 + x];
        float c = 0.f;
        #pragma unroll
        for (int i = 0; i < 16; ++i) c += src[i] * fc_b2[off + i * 16 + o];
        Cvec[col] = c;
        for (int k = 0; k < 32; ++k) {
            const float* w2 = fc_w2 + (size_t)k * 1024 + off + o;
            float a = 0.f;
            #pragma unroll
            for (int i = 0; i < 16; ++i) a += src[i] * w2[i * 16];
            Mt[col * 32 + (((k >> 3) & 1) << 4) + ((k >> 4) << 3) + (k & 7)] = (_Float16)a;
        }
    }
    __syncthreads();

    // ---------------- Phase B: per-wave WMMA over m-tiles ----------------
    const int wave  = tid >> 5;
    const int lane  = tid & 31;
    const int lmod  = lane & 15;
    const int lhalf = lane >> 4;

    // B fragments of fused matrix: element(k,ncol) -> lane = ncol%16 + 16*(k/8%2), half = 8*(k/16) + k%8
    v16h bm[8];
    #pragma unroll
    for (int j = 0; j < 8; ++j) bm[j] = lds_load16h(&Mt[(lmod + 16 * j) * 32 + 16 * lhalf]);

    // A fragments of fc_w1^T (for GEMM1': D' = w1^T x ea^T = h^T).
    // A'0 row krow=lmod, A'1 row krow=16+lmod; element(krow,j) = fc_w1[j*32+krow].
    // A layout: halves 0-7 -> j = 8*lhalf + i ; halves 8-15 -> j = 16 + 8*lhalf + i.
    v16h aw1lo, aw1hi;
    #pragma unroll
    for (int i = 0; i < 8; ++i) {
        aw1lo[i]     = (_Float16)fc_w1[(8 * lhalf + i) * 32 + lmod];
        aw1lo[i + 8] = (_Float16)fc_w1[(16 + 8 * lhalf + i) * 32 + lmod];
        aw1hi[i]     = (_Float16)fc_w1[(8 * lhalf + i) * 32 + 16 + lmod];
        aw1hi[i + 8] = (_Float16)fc_w1[(16 + 8 * lhalf + i) * 32 + 16 + lmod];
    }

    // loop-invariant fc_b1 biases: d0 krow = r + 8*lhalf ; d1 krow = 16 + r + 8*lhalf
    float fb1lo[8], fb1hi[8];
    #pragma unroll
    for (int r = 0; r < 8; ++r) {
        fb1lo[r] = fb1_sh[8 * lhalf + r];
        fb1hi[r] = fb1_sh[16 + 8 * lhalf + r];
    }

    const float cs   = Cvec[lmod];
    const float csv  = Cvec[16 + lmod];
    const float cvs0 = Cvec[32 + lmod], cvs1 = Cvec[48 + lmod], cvs2 = Cvec[64 + lmod];
    const float cbv0 = Cvec[80 + lmod], cbv1 = Cvec[96 + lmod], cbv2 = Cvec[112 + lmod];

    float accS = 0.f, accV0 = 0.f, accV1 = 0.f, accV2 = 0.f;
    const float inv_fan = 0.17677669529663687f;   // 1/sqrt(C0+C1)
    const float rsqrt3  = 0.57735026918962576f;

    const v8f zacc = {0.f, 0.f, 0.f, 0.f, 0.f, 0.f, 0.f, 0.f};
    const int ntiles = N >> 4;

    for (int t = wave; t < ntiles; t += 4) {
        const int m0 = t * 16;

        // edge_attr tile as B fragment of GEMM1': element(j, m) = ea[m0+m][j]
        // lane: m = lmod, half<8: j = 8*lhalf + half ; half>=8: j = 16 + 8*lhalf + (half-8)
        const float* rp = edge_attr + ((size_t)node * N + (m0 + lmod)) * NEF + lhalf * 8;
        float4 x0 = *(const float4*)(rp);
        float4 x1 = *(const float4*)(rp + 4);
        float4 x2 = *(const float4*)(rp + 16);
        float4 x3 = *(const float4*)(rp + 20);
        v16h bE;
        bE[0]=(_Float16)x0.x;  bE[1]=(_Float16)x0.y;  bE[2]=(_Float16)x0.z;  bE[3]=(_Float16)x0.w;
        bE[4]=(_Float16)x1.x;  bE[5]=(_Float16)x1.y;  bE[6]=(_Float16)x1.z;  bE[7]=(_Float16)x1.w;
        bE[8]=(_Float16)x2.x;  bE[9]=(_Float16)x2.y;  bE[10]=(_Float16)x2.z; bE[11]=(_Float16)x2.w;
        bE[12]=(_Float16)x3.x; bE[13]=(_Float16)x3.y; bE[14]=(_Float16)x3.z; bE[15]=(_Float16)x3.w;

        // GEMM1': h^T tiles. d0: krows 0-15, d1: krows 16-31. D'(krow,m): lane m=lmod, vgpr r -> krow=r+8*lhalf
        v8f d0 = __builtin_amdgcn_wmma_f32_16x16x32_f16(false, aw1lo, false, bE, (short)0, zacc, false, false);
        v8f d1 = __builtin_amdgcn_wmma_f32_16x16x32_f16(false, aw1hi, false, bE, (short)0, zacc, false, false);

        // bias + relu + cvt, entirely in registers: D' layout == A layout of GEMM2
        v16h ah;
        #pragma unroll
        for (int r = 0; r < 8; ++r) {
            float h0 = d0[r] + fb1lo[r]; h0 = h0 > 0.f ? h0 : 0.f;
            float h1 = d1[r] + fb1hi[r]; h1 = h1 > 0.f ? h1 : 0.f;
            ah[r]     = (_Float16)h0;   // K = 8*lhalf + r
            ah[r + 8] = (_Float16)h1;   // K = 16 + 8*lhalf + r
        }

        // GEMM2: y = h @ M  (16x128)
        v8f y[8];
        #pragma unroll
        for (int j = 0; j < 8; ++j)
            y[j] = __builtin_amdgcn_wmma_f32_16x16x32_f16(false, ah, false, bm[j], (short)0, zacc, false, false);

        // epilogue: combine with edge_sh, mask, accumulate over m
        #pragma unroll
        for (int r = 0; r < 8; ++r) {
            const int m = m0 + r + 8 * lhalf;
            float4 esh = *(const float4*)(edge_sh + ((size_t)node * N + m) * 4);
            float mv = mask[(size_t)b * N + m];
            float mfac = (m == n) ? 0.f : mv;
            float shs = esh.x, sv0 = esh.y, sv1 = esh.z, sv2 = esh.w;
            float tss = y[0][r] + cs;
            float tsv = y[1][r] + csv;
            float tvs0 = y[2][r] + cvs0, tvs1 = y[3][r] + cvs1, tvs2 = y[4][r] + cvs2;
            float hb0 = y[5][r] + cbv0, hb1 = y[6][r] + cbv1, hb2 = y[7][r] + cbv2;
            float tvv = (hb0 * sv0 + hb1 * sv1 + hb2 * sv2) * rsqrt3;
            float g = inv_fan * mfac;
            accS  += (tss * shs + tvv) * g;
            accV0 += (tsv * sv0 + tvs0 * shs) * g;
            accV1 += (tsv * sv1 + tvs1 * shs) * g;
            accV2 += (tsv * sv2 + tvs2 * shs) * g;
        }
    }

    // ---------------- Phase C: reduce + output linear + residual ----------------
    redS[tid] = accS;
    redV[tid * 3 + 0] = accV0;
    redV[tid * 3 + 1] = accV1;
    redV[tid * 3 + 2] = accV2;
    __syncthreads();

    if (tid < 16) {
        float inv_d = 1.0f / dsum_sh;
        float s = 0.f, v0 = 0.f, v1 = 0.f, v2 = 0.f;
        #pragma unroll
        for (int w = 0; w < 4; ++w) {
            int i0 = w * 32 + tid, i1 = i0 + 16;
            s  += redS[i0] + redS[i1];
            v0 += redV[i0 * 3 + 0] + redV[i1 * 3 + 0];
            v1 += redV[i0 * 3 + 1] + redV[i1 * 3 + 1];
            v2 += redV[i0 * 3 + 2] + redV[i1 * 3 + 2];
        }
        Ssum[tid] = s * inv_d;
        Vsum[tid * 3 + 0] = v0 * inv_d;
        Vsum[tid * 3 + 1] = v1 * inv_d;
        Vsum[tid * 3 + 2] = v2 * inv_d;
    }
    __syncthreads();

    if (tid < 16) {
        float a = 0.f;
        #pragma unroll
        for (int o = 0; o < 16; ++o) a += Ssum[o] * w_out_s[o * 16 + tid];
        out[(size_t)node * DIN + tid] = a * 0.25f + na_sh[tid];
    } else if (tid < 64) {
        int idx = tid - 16; int o2 = idx / 3; int x = idx % 3;
        float a = 0.f;
        #pragma unroll
        for (int o = 0; o < 16; ++o) a += Vsum[o * 3 + x] * w_out_v[o * 16 + o2];
        out[(size_t)node * DIN + 16 + o2 * 3 + x] = a * 0.25f + na_sh[16 + o2 * 3 + x];
    }
}

extern "C" void kernel_launch(void* const* d_in, const int* in_sizes, int n_in,
                              void* d_out, int out_size, void* d_ws, size_t ws_size,
                              hipStream_t stream) {
    const float* node_attr = (const float*)d_in[0];
    const float* edge_attr = (const float*)d_in[1];
    const float* edge_sh   = (const float*)d_in[2];
    const float* mask      = (const float*)d_in[3];
    const float* w_in_s    = (const float*)d_in[4];
    const float* w_in_v    = (const float*)d_in[5];
    const float* fc_w1     = (const float*)d_in[6];
    const float* fc_b1     = (const float*)d_in[7];
    const float* fc_w2     = (const float*)d_in[8];
    const float* fc_b2     = (const float*)d_in[9];
    const float* w_out_s   = (const float*)d_in[10];
    const float* w_out_v   = (const float*)d_in[11];
    float* out = (float*)d_out;

    const int nodes_total = in_sizes[3];                 // B*N (mask element count)
    const int N = in_sizes[1] / (nodes_total * NEF);     // edge_attr = B*N*N*32

    tpconv_fused_kernel<<<dim3(nodes_total), dim3(128), 0, stream>>>(
        node_attr, edge_attr, edge_sh, mask,
        w_in_s, w_in_v, fc_w1, fc_b1, fc_w2, fc_b2,
        w_out_s, w_out_v, out, N);
}
